// S4Model_20624432956218
// MI455X (gfx1250) — compile-verified, matrix operand order checked by reading, not run on hardware
//
#include <hip/hip_runtime.h>
#include <hip/hip_bf16.h>
#include <math.h>

#define B_ 16
#define L_ 4096
#define DIN_ 16
#define H_ 256
#define N2_ 32
#define NL_ 4
#define TCH 128
#define NCH (L_ / TCH)   // 32 chunks
#define LN_EPS 1e-5f

typedef __attribute__((ext_vector_type(16))) __bf16        v16bf;
typedef __attribute__((ext_vector_type(8)))  float         v8f;
typedef __attribute__((ext_vector_type(8)))  unsigned int  v8u;

// float -> bf16 with round-to-nearest-even
static __device__ __forceinline__ unsigned short f2bf(float f) {
  unsigned int u = __builtin_bit_cast(unsigned int, f);
  unsigned int r = u + 0x7FFFu + ((u >> 16) & 1u);
  return (unsigned short)(r >> 16);
}

// Load a 16x32 bf16 fragment row from LDS (row stored as 32 contiguous bf16).
// Per ISA 7.12.2: lanes 0-15 / 16-31 split K halves; dword v holds
// K = ((v>>2)<<4) + 8*g + 2*(v&3) .. +1  (g = lane>=16). K-pairs are
// contiguous in our [row][K] LDS layout -> one b32 load per dword.
static __device__ __forceinline__ v16bf frag_ld(const unsigned short* row_ptr, int lane) {
  int g = (lane >> 4) & 1;
  const unsigned int* p = (const unsigned int*)row_ptr;
  v8u r;
#pragma unroll
  for (int v = 0; v < 8; ++v) {
    int wi = ((v >> 2) << 3) | (g << 2) | (v & 3);
    r[v] = p[wi];
  }
  return __builtin_bit_cast(v16bf, r);
}

// ---------------- encoder: u[b,h,l] = x[b,l,:] @ enc_w + enc_b ----------------
__global__ __launch_bounds__(256) void enc_kernel(const float* __restrict__ x,
    const float* __restrict__ w, const float* __restrict__ bias,
    float* __restrict__ u) {
  __shared__ float sx[32][DIN_];
  __shared__ float sw[DIN_][H_];
  int bid = blockIdx.x;
  int b = bid >> 7;               // L/32 = 128 tiles
  int l0 = (bid & 127) << 5;
  int tid = threadIdx.x;
  if (tid < 128) {                // 32 l * 4 float4
    int l = tid >> 2, d4 = tid & 3;
    *(float4*)&sx[l][d4 * 4] =
        *(const float4*)(x + ((size_t)(b * L_ + l0 + l)) * DIN_ + d4 * 4);
  }
#pragma unroll
  for (int it = 0; it < 4; ++it) {
    int e = tid + it * 256;
    int d = e >> 6, h4 = e & 63;
    *(float4*)&sw[d][h4 * 4] = *(const float4*)(w + d * H_ + h4 * 4);
  }
  __syncthreads();
  int l = tid & 31, hg = tid >> 5;
#pragma unroll 4
  for (int hh = 0; hh < 32; ++hh) {
    int h = hg * 32 + hh;
    float acc = bias[h];
#pragma unroll
    for (int d = 0; d < DIN_; ++d) acc += sx[l][d] * sw[d][h];
    u[((size_t)(b * H_ + h)) * L_ + l0 + l] = acc;
  }
}

// ------------- per-layer S4D discretization tables (lambda, 2*C*(lam-1)/A) ----
__global__ __launch_bounds__(256) void table_kernel(const float* __restrict__ log_dt,
    const float* __restrict__ C_re, const float* __restrict__ C_im,
    const float* __restrict__ logA_re, const float* __restrict__ A_im,
    float* __restrict__ lamr, float* __restrict__ lami,
    float* __restrict__ ccr, float* __restrict__ cci) {
  int i = blockIdx.x * 256 + threadIdx.x;   // i < H*N2
  int h = i >> 5;
  float dt  = expf(log_dt[h]);
  float Are = -expf(logA_re[i]);
  float Aim = A_im[i];
  float dr = dt * Are, di = dt * Aim;
  float em = expf(dr);
  float lr = em * cosf(di), li = em * sinf(di);
  float inv = 1.0f / (Are * Are + Aim * Aim);
  float fr = ((lr - 1.0f) * Are + li * Aim) * inv;
  float fi = (li * Are - (lr - 1.0f) * Aim) * inv;
  float cr = C_re[i], ci = C_im[i];
  lamr[i] = lr; lami[i] = li;
  ccr[i] = 2.0f * (cr * fr - ci * fi);
  cci[i] = 2.0f * (cr * fi + ci * fr);
}

// ---------------- layernorm over channel dim (u is [B,H,L]) -------------------
__global__ __launch_bounds__(256) void ln_kernel(const float* __restrict__ u,
    float* __restrict__ z, const float* __restrict__ nw, const float* __restrict__ nb) {
  __shared__ float st[H_ * 33];
  __shared__ float ps[8 * 32], pq[8 * 32];
  __shared__ float smean[32], srstd[32];
  int bid = blockIdx.x;
  int b = bid >> 7;
  int l0 = (bid & 127) << 5;
  int tid = threadIdx.x;
  int lq = tid & 7, hs = tid >> 3;            // 8 l-quads x 32 h
#pragma unroll
  for (int r = 0; r < 8; ++r) {
    int h = r * 32 + hs;
    float4 v = *(const float4*)(u + ((size_t)(b * H_ + h)) * L_ + l0 + lq * 4);
    st[h * 33 + lq * 4 + 0] = v.x;
    st[h * 33 + lq * 4 + 1] = v.y;
    st[h * 33 + lq * 4 + 2] = v.z;
    st[h * 33 + lq * 4 + 3] = v.w;
  }
  __syncthreads();
  int l = tid & 31, g = tid >> 5;
  float s = 0.f, q = 0.f;
  for (int r = 0; r < 32; ++r) {
    float v = st[(g * 32 + r) * 33 + l];
    s += v; q += v * v;
  }
  ps[g * 32 + l] = s; pq[g * 32 + l] = q;
  __syncthreads();
  if (tid < 32) {
    float ss = 0.f, qq = 0.f;
    for (int p = 0; p < 8; ++p) { ss += ps[p * 32 + tid]; qq += pq[p * 32 + tid]; }
    float m = ss * (1.0f / H_);
    float var = qq * (1.0f / H_) - m * m;
    smean[tid] = m; srstd[tid] = rsqrtf(var + LN_EPS);
  }
  __syncthreads();
#pragma unroll
  for (int r = 0; r < 8; ++r) {
    int h = r * 32 + hs;
    float wgt = nw[h], bgt = nb[h];
    float4 o;
    o.x = (st[h * 33 + lq * 4 + 0] - smean[lq * 4 + 0]) * srstd[lq * 4 + 0] * wgt + bgt;
    o.y = (st[h * 33 + lq * 4 + 1] - smean[lq * 4 + 1]) * srstd[lq * 4 + 1] * wgt + bgt;
    o.z = (st[h * 33 + lq * 4 + 2] - smean[lq * 4 + 2]) * srstd[lq * 4 + 2] * wgt + bgt;
    o.w = (st[h * 33 + lq * 4 + 3] - smean[lq * 4 + 3]) * srstd[lq * 4 + 3] * wgt + bgt;
    *(float4*)(z + ((size_t)(b * H_ + h)) * L_ + l0 + lq * 4) = o;
  }
}

// ------------- pass A: chunk-local diagonal-state recurrence ------------------
__global__ __launch_bounds__(256) void scan_kernel(const float* __restrict__ z,
    float* __restrict__ y,
    const float* __restrict__ lamr_t, const float* __restrict__ lami_t,
    const float* __restrict__ ccr_t, const float* __restrict__ cci_t,
    float2* __restrict__ states) {
  int idx = blockIdx.x * 256 + threadIdx.x;   // over B*H*NCH
  int bh = idx / NCH;
  int c  = idx & (NCH - 1);
  int h  = bh & (H_ - 1);
  float lr[N2_], li[N2_], cr[N2_], ci[N2_], sr[N2_], si[N2_];
#pragma unroll
  for (int n = 0; n < N2_; ++n) {
    lr[n] = lamr_t[h * N2_ + n]; li[n] = lami_t[h * N2_ + n];
    cr[n] = ccr_t[h * N2_ + n];  ci[n] = cci_t[h * N2_ + n];
    sr[n] = 0.f; si[n] = 0.f;
  }
  const float* up = z + (size_t)bh * L_ + c * TCH;
  float* yp = y + (size_t)bh * L_ + c * TCH;
  float4 uv = *(const float4*)(up);           // software pipeline: stream load ahead
  for (int t = 0; t < TCH; t += 4) {
    if ((t & 31) == 0) __builtin_prefetch(up + t + 64, 0, 0);  // global_prefetch_b8
    float4 nxt = uv;
    if (t + 4 < TCH) nxt = *(const float4*)(up + t + 4);
    float ui4[4] = {uv.x, uv.y, uv.z, uv.w};
    float oo[4];
#pragma unroll
    for (int j = 0; j < 4; ++j) {
      float ut = ui4[j];
      float acc = 0.f;
#pragma unroll
      for (int n = 0; n < N2_; ++n) {
        float nr = lr[n] * sr[n] - li[n] * si[n] + ut;
        float ni = lr[n] * si[n] + li[n] * sr[n];
        sr[n] = nr; si[n] = ni;
        acc += cr[n] * nr - ci[n] * ni;
      }
      oo[j] = acc;
    }
    *(float4*)(yp + t) = make_float4(oo[0], oo[1], oo[2], oo[3]);
    uv = nxt;
  }
  float2* sp = states + (size_t)idx * N2_;
#pragma unroll
  for (int n = 0; n < N2_; ++n) sp[n] = make_float2(sr[n], si[n]);
}

// ------------- pass B: sequential carry over chunks (lam^T via squaring) ------
__global__ __launch_bounds__(256) void carry_kernel(float2* __restrict__ states,
    const float* __restrict__ lamr_t, const float* __restrict__ lami_t) {
  int idx = blockIdx.x * 256 + threadIdx.x;   // over B*H*N2
  int bh = idx / N2_;
  int n  = idx & (N2_ - 1);
  int h  = bh & (H_ - 1);
  float pr = lamr_t[h * N2_ + n], pi = lami_t[h * N2_ + n];
#pragma unroll
  for (int s = 0; s < 7; ++s) {               // TCH = 128 = 2^7
    float nr = pr * pr - pi * pi;
    float ni = 2.f * pr * pi;
    pr = nr; pi = ni;
  }
  float car = 0.f, cai = 0.f;
  for (int c = 0; c < NCH; ++c) {
    float2* p = states + ((size_t)(bh * NCH + c)) * N2_ + n;
    float2 loc = *p;
    *p = make_float2(car, cai);               // carry-in for chunk c
    float nr = pr * car - pi * cai + loc.x;
    float ni = pr * cai + pi * car + loc.y;
    car = nr; cai = ni;
  }
}

// --- pass C: carry fixup + D skip + exact GELU; emits bf16 for the WMMA GEMM --
__global__ __launch_bounds__(256) void fixup_kernel(const float* __restrict__ y,
    const float* __restrict__ z,
    const float* __restrict__ lamr_t, const float* __restrict__ lami_t,
    const float* __restrict__ ccr_t, const float* __restrict__ cci_t,
    const float2* __restrict__ states, const float* __restrict__ Dv,
    unsigned short* __restrict__ ybf) {
  int idx = blockIdx.x * 256 + threadIdx.x;   // over B*H*NCH
  int bh = idx / NCH;
  int c  = idx & (NCH - 1);
  int h  = bh & (H_ - 1);
  float dskip = Dv[h];
  float lr[N2_], li[N2_], cr[N2_], ci[N2_], gr[N2_], gi[N2_];
  const float2* sp = states + (size_t)idx * N2_;
#pragma unroll
  for (int n = 0; n < N2_; ++n) {
    lr[n] = lamr_t[h * N2_ + n]; li[n] = lami_t[h * N2_ + n];
    cr[n] = ccr_t[h * N2_ + n];  ci[n] = cci_t[h * N2_ + n];
    float2 s0 = sp[n];
    gr[n] = lr[n] * s0.x - li[n] * s0.y;      // lam^(t+1) * carry, t = 0
    gi[n] = lr[n] * s0.y + li[n] * s0.x;
  }
  const float* up = z + (size_t)bh * L_ + c * TCH;
  const float* yp = y + (size_t)bh * L_ + c * TCH;
  unsigned short* op = ybf + (size_t)bh * L_ + c * TCH;
  float4 yv = *(const float4*)(yp);
  float4 zv = *(const float4*)(up);
  for (int t = 0; t < TCH; t += 4) {
    float4 yn = yv, zn = zv;
    if (t + 4 < TCH) {
      yn = *(const float4*)(yp + t + 4);
      zn = *(const float4*)(up + t + 4);
    }
    float yi4[4] = {yv.x, yv.y, yv.z, yv.w};
    float zi4[4] = {zv.x, zv.y, zv.z, zv.w};
    unsigned short ob[4];
#pragma unroll
    for (int j = 0; j < 4; ++j) {
      float corr = 0.f;
#pragma unroll
      for (int n = 0; n < N2_; ++n) {
        corr += cr[n] * gr[n] - ci[n] * gi[n];
        float nr = lr[n] * gr[n] - li[n] * gi[n];
        float ni = lr[n] * gi[n] + li[n] * gr[n];
        gr[n] = nr; gi[n] = ni;
      }
      float v = yi4[j] + corr + dskip * zi4[j];
      ob[j] = f2bf(0.5f * v * (1.0f + erff(v * 0.70710678118654752440f)));
    }
    uint2 pk;
    pk.x = (unsigned)ob[0] | ((unsigned)ob[1] << 16);
    pk.y = (unsigned)ob[2] | ((unsigned)ob[3] << 16);
    *(uint2*)(op + t) = pk;
    yv = yn; zv = zn;
  }
}

// ------------- output projection (2H x H) + GLU + residual, WMMA bf16 --------
__global__ __launch_bounds__(256) void outproj_kernel(const unsigned short* __restrict__ ybf,
    const float* __restrict__ w, const float* __restrict__ bias,
    float* __restrict__ u) {
  __shared__ unsigned short sy[64 * 32];      // [l][k] bf16, 4 KB
  __shared__ unsigned short sw2[512 * 32];    // [o][k] bf16, 32 KB
  int bid = blockIdx.x;
  int b = bid >> 6;              // L/64 = 64 l-tiles
  int l0 = (bid & 63) << 6;
  int tid = threadIdx.x;
  int lane = tid & 31, wv = tid >> 5;
  v8f accA[2][4], accG[2][4];
#pragma unroll
  for (int mi = 0; mi < 2; ++mi)
#pragma unroll
    for (int ni = 0; ni < 4; ++ni)
#pragma unroll
      for (int r = 0; r < 8; ++r) { accA[mi][ni][r] = 0.f; accG[mi][ni][r] = 0.f; }

  for (int kk = 0; kk < H_ / 32; ++kk) {
    int k0 = kk * 32;
    // y tile: bf16 in global already; b64 loads of 4 l (fixed k). Fully
    // unrolled so all tile loads are in flight under one batched wait.
#pragma unroll
    for (int it = 0; it < 2; ++it) {
      int e = tid + it * 256;
      int k = e >> 4, l4 = e & 15;
      uint2 v = *(const uint2*)(ybf + ((size_t)(b * H_ + k0 + k)) * L_ + l0 + l4 * 4);
      sy[(l4 * 4 + 0) * 32 + k] = (unsigned short)(v.x & 0xffffu);
      sy[(l4 * 4 + 1) * 32 + k] = (unsigned short)(v.x >> 16);
      sy[(l4 * 4 + 2) * 32 + k] = (unsigned short)(v.y & 0xffffu);
      sy[(l4 * 4 + 3) * 32 + k] = (unsigned short)(v.y >> 16);
    }
    // weight tile: float4 along k, pack to bf16, one b64 LDS store
#pragma unroll
    for (int it = 0; it < 16; ++it) {
      int e = tid + it * 256;
      int o = e >> 3, k4 = e & 7;
      float4 wv4 = *(const float4*)(w + (size_t)o * H_ + k0 + k4 * 4);
      uint2 pk;
      pk.x = (unsigned)f2bf(wv4.x) | ((unsigned)f2bf(wv4.y) << 16);
      pk.y = (unsigned)f2bf(wv4.z) | ((unsigned)f2bf(wv4.w) << 16);
      *(uint2*)&sw2[o * 32 + k4 * 4] = pk;
    }
    __syncthreads();
    v16bf bf[4];
#pragma unroll
    for (int ni = 0; ni < 4; ++ni)
      bf[ni] = frag_ld(&sy[(ni * 16 + (lane & 15)) * 32], lane);
#pragma unroll
    for (int mi = 0; mi < 2; ++mi) {
      int mt = 2 * wv + mi;
      v16bf aA = frag_ld(&sw2[(mt * 16 + (lane & 15)) * 32], lane);
      v16bf aG = frag_ld(&sw2[((mt + 16) * 16 + (lane & 15)) * 32], lane);
#pragma unroll
      for (int ni = 0; ni < 4; ++ni) {
        accA[mi][ni] = __builtin_amdgcn_wmma_f32_16x16x32_bf16(
            false, aA, false, bf[ni], (short)0, accA[mi][ni], false, false);
        accG[mi][ni] = __builtin_amdgcn_wmma_f32_16x16x32_bf16(
            false, aG, false, bf[ni], (short)0, accG[mi][ni], false, false);
      }
    }
    __syncthreads();
  }
  // epilogue: GLU + residual.  C layout: m = r + 8*(lane>=16), n = lane&15
  int n = lane & 15;
  int mo = (lane >> 4) << 3;
#pragma unroll
  for (int mi = 0; mi < 2; ++mi) {
    int mt = 2 * wv + mi;
#pragma unroll
    for (int ni = 0; ni < 4; ++ni) {
#pragma unroll
      for (int r = 0; r < 8; ++r) {
        int oa = mt * 16 + r + mo;
        int lcol = l0 + ni * 16 + n;
        float av = accA[mi][ni][r] + bias[oa];
        float gv = accG[mi][ni][r] + bias[oa + H_];
        float glu = av / (1.0f + expf(-gv));
        size_t ui = ((size_t)(b * H_ + oa)) * L_ + lcol;
        u[ui] = glu + u[ui];
      }
    }
  }
}

// ------------- decoder: out[b,l,:] = u[b,:,l]^T @ dec_w + dec_b, WMMA --------
__global__ __launch_bounds__(256) void dec_kernel(const float* __restrict__ u,
    const float* __restrict__ w, const float* __restrict__ bias,
    float* __restrict__ out) {
  __shared__ unsigned short sa[128 * 32];     // [l][k] bf16, 8 KB
  __shared__ unsigned short sb[256 * 32];     // [j][k] bf16, 16 KB
  int bid = blockIdx.x;
  int b = bid >> 5;              // L/128 = 32 l-tiles
  int l0 = (bid & 31) << 7;
  int tid = threadIdx.x;
  int lane = tid & 31, wv = tid >> 5;
  v8f acc[16];
#pragma unroll
  for (int nt = 0; nt < 16; ++nt)
#pragma unroll
    for (int r = 0; r < 8; ++r) acc[nt][r] = 0.f;

  for (int kk = 0; kk < H_ / 32; ++kk) {
    int k0 = kk * 32;
    // u^T tile: float4 along l, 4 b16 stores (fully unrolled -> batched loads)
#pragma unroll
    for (int it = 0; it < 4; ++it) {
      int e = tid + it * 256;
      int k = e >> 5, l4 = e & 31;
      float4 uv = *(const float4*)(u + ((size_t)(b * H_ + k0 + k)) * L_ + l0 + l4 * 4);
      sa[(l4 * 4 + 0) * 32 + k] = f2bf(uv.x);
      sa[(l4 * 4 + 1) * 32 + k] = f2bf(uv.y);
      sa[(l4 * 4 + 2) * 32 + k] = f2bf(uv.z);
      sa[(l4 * 4 + 3) * 32 + k] = f2bf(uv.w);
    }
    // dec_w tile: float4 along j, 4 b16 stores
#pragma unroll
    for (int it = 0; it < 8; ++it) {
      int e = tid + it * 256;
      int k = e >> 6, j4 = e & 63;
      float4 wv4 = *(const float4*)(w + (size_t)(k0 + k) * H_ + j4 * 4);
      sb[(j4 * 4 + 0) * 32 + k] = f2bf(wv4.x);
      sb[(j4 * 4 + 1) * 32 + k] = f2bf(wv4.y);
      sb[(j4 * 4 + 2) * 32 + k] = f2bf(wv4.z);
      sb[(j4 * 4 + 3) * 32 + k] = f2bf(wv4.w);
    }
    __syncthreads();
    v16bf af = frag_ld(&sa[(wv * 16 + (lane & 15)) * 32], lane);
#pragma unroll
    for (int nt = 0; nt < 16; ++nt) {
      v16bf bfr = frag_ld(&sb[(nt * 16 + (lane & 15)) * 32], lane);
      acc[nt] = __builtin_amdgcn_wmma_f32_16x16x32_bf16(
          false, af, false, bfr, (short)0, acc[nt], false, false);
    }
    __syncthreads();
  }
  int n = lane & 15;
  int mo = (lane >> 4) << 3;
#pragma unroll
  for (int nt = 0; nt < 16; ++nt) {
#pragma unroll
    for (int r = 0; r < 8; ++r) {
      int lrow = l0 + wv * 16 + r + mo;
      int j = nt * 16 + n;
      out[((size_t)(b * L_ + lrow)) * H_ + j] = acc[nt][r] + bias[j];
    }
  }
}

extern "C" void kernel_launch(void* const* d_in, const int* in_sizes, int n_in,
                              void* d_out, int out_size, void* d_ws, size_t ws_size,
                              hipStream_t stream) {
  (void)in_sizes; (void)n_in; (void)out_size; (void)ws_size;
  const float* x      = (const float*)d_in[0];
  const float* enc_w  = (const float*)d_in[1];
  const float* enc_b  = (const float*)d_in[2];
  const float* log_dt = (const float*)d_in[3];
  const float* C_re   = (const float*)d_in[4];
  const float* C_im   = (const float*)d_in[5];
  const float* logAre = (const float*)d_in[6];
  const float* A_im   = (const float*)d_in[7];
  const float* Dv     = (const float*)d_in[8];
  const float* norm_w = (const float*)d_in[9];
  const float* norm_b = (const float*)d_in[10];
  const float* out_w  = (const float*)d_in[11];
  const float* out_b  = (const float*)d_in[12];
  const float* dec_w  = (const float*)d_in[13];
  const float* dec_b  = (const float*)d_in[14];

  char* ws = (char*)d_ws;
  const size_t SZ   = (size_t)B_ * H_ * L_ * sizeof(float);            // 64 MB
  const size_t BSZ  = (size_t)B_ * H_ * L_ * sizeof(unsigned short);   // 32 MB
  const size_t STSZ = (size_t)B_ * H_ * NCH * N2_ * sizeof(float2);    // 32 MB
  float*          u      = (float*)(ws);
  float*          zbuf   = (float*)(ws + SZ);
  float*          ybuf   = (float*)(ws + 2 * SZ);
  unsigned short* ybf    = (unsigned short*)(ws + 3 * SZ);
  float2*         states = (float2*)(ws + 3 * SZ + BSZ);
  float*          lamr   = (float*)(ws + 3 * SZ + BSZ + STSZ);
  float*          lami   = lamr + H_ * N2_;
  float*          ccr    = lami + H_ * N2_;
  float*          cci    = ccr  + H_ * N2_;

  enc_kernel<<<B_ * (L_ / 32), 256, 0, stream>>>(x, enc_w, enc_b, u);
  for (int layer = 0; layer < NL_; ++layer) {
    table_kernel<<<(H_ * N2_) / 256, 256, 0, stream>>>(
        log_dt + layer * H_,
        C_re + (size_t)layer * H_ * N2_, C_im + (size_t)layer * H_ * N2_,
        logAre + (size_t)layer * H_ * N2_, A_im + (size_t)layer * H_ * N2_,
        lamr, lami, ccr, cci);
    ln_kernel<<<B_ * (L_ / 32), 256, 0, stream>>>(
        u, zbuf, norm_w + layer * H_, norm_b + layer * H_);
    scan_kernel<<<(B_ * H_ * NCH) / 256, 256, 0, stream>>>(
        zbuf, ybuf, lamr, lami, ccr, cci, states);
    carry_kernel<<<(B_ * H_ * N2_) / 256, 256, 0, stream>>>(states, lamr, lami);
    fixup_kernel<<<(B_ * H_ * NCH) / 256, 256, 0, stream>>>(
        ybuf, zbuf, lamr, lami, ccr, cci, states, Dv + layer * H_, ybf);
    outproj_kernel<<<B_ * (L_ / 64), 256, 0, stream>>>(
        ybf, out_w + (size_t)layer * 2 * H_ * H_, out_b + (size_t)layer * 2 * H_, u);
  }
  dec_kernel<<<B_ * (L_ / 128), 256, 0, stream>>>(u, dec_w, dec_b, (float*)d_out);
}
